// CrossAttention_7172595384967
// MI455X (gfx1250) — matrix operand order repaired and checked by reading, not compiled, not run on hardware
//
#include <hip/hip_runtime.h>

// ---------------------------------------------------------------------------
// CDNA5 (gfx1250, wave32) cross-attention pipeline, bf16 WMMA everywhere.
// GEMMs use double-buffered GLOBAL_LOAD_ASYNC_TO_LDS staging when available.
// ---------------------------------------------------------------------------

typedef __attribute__((ext_vector_type(16))) __bf16 v16bf;
typedef __attribute__((ext_vector_type(8)))  float  v8f;
typedef unsigned short u16;
typedef unsigned int   u32;

constexpr int B_  = 2;
constexpr int S_  = 4096;
constexpr int P_  = 1024;
constexpr int D_  = 1024;
constexpr int H_  = 16;
constexpr int DH_ = 64;

union Frag  { v16bf v; uint4 q[2]; };
union Pack8 { u16 s[8]; uint4 q; };

__device__ __forceinline__ u16 f2bf(float f) {
  u32 u = __float_as_uint(f);
  u += 0x7FFFu + ((u >> 16) & 1u);          // round-to-nearest-even
  return (u16)(u >> 16);
}

__device__ __forceinline__ v8f wmma_bf16(v16bf a, v16bf b, v8f c) {
  // D(f32 16x16) = A(bf16 16x32) * B(bf16 32x16) + C
  return __builtin_amdgcn_wmma_f32_16x16x32_bf16(
      /*neg_a=*/false, a, /*neg_b=*/false, b,
      /*c_mod=*/(short)0, c, /*reuse_a=*/false, /*reuse_b=*/false);
}

// ---- gfx1250 async global->LDS copy (ASYNCcnt path), with safe fallback ----
typedef int v4i __attribute__((vector_size(16)));
typedef __attribute__((address_space(1))) v4i as1_v4i;   // global
typedef __attribute__((address_space(3))) v4i as3_v4i;   // LDS

__device__ __forceinline__ void async_g2l_b128(const u16* g, u16* l) {
#if __has_builtin(__builtin_amdgcn_global_load_async_to_lds_b128)
  __builtin_amdgcn_global_load_async_to_lds_b128((as1_v4i*)g, (as3_v4i*)l, 0, 0);
#else
  *(uint4*)l = *(const uint4*)g;            // sync fallback (same barriers)
#endif
}

template<int N>
__device__ __forceinline__ void wait_async() {
#if __has_builtin(__builtin_amdgcn_s_wait_asynccnt)
  __builtin_amdgcn_s_wait_asynccnt((short)N);
#elif __has_builtin(__builtin_amdgcn_global_load_async_to_lds_b128)
  asm volatile("s_wait_asynccnt %0" :: "i"(N));
#else
  __threadfence_block();
#endif
}

// ---------------------------------------------------------------------------
// Kernel 1: inclusive cumsum of patch boundaries along S (per batch row).
// ---------------------------------------------------------------------------
__global__ __launch_bounds__(256) void cumsum_kernel(const int* __restrict__ bnd,
                                                     int* __restrict__ pidx) {
  __shared__ int tot[256];
  constexpr int CH = S_ / 256;   // 16
  const int b = blockIdx.x;
  const int t = threadIdx.x;
  const int* src = bnd + (size_t)b * S_;
  int* dst = pidx + (size_t)b * S_;

  int loc[CH];
  int s = 0;
  #pragma unroll
  for (int i = 0; i < CH; ++i) { s += src[t * CH + i]; loc[i] = s; }
  tot[t] = s;
  __syncthreads();
  for (int off = 1; off < 256; off <<= 1) {
    int v = (t >= off) ? tot[t - off] : 0;
    __syncthreads();
    tot[t] += v;
    __syncthreads();
  }
  const int pre = (t == 0) ? 0 : tot[t - 1];
  #pragma unroll
  for (int i = 0; i < CH; ++i) dst[t * CH + i] = loc[i] + pre;
}

// ---------------------------------------------------------------------------
// Kernel 2: f32 -> bf16 conversion (vectorized x4).
// ---------------------------------------------------------------------------
__global__ void cvt_bf16(const float* __restrict__ src, u16* __restrict__ dst, int n4) {
  int i = blockIdx.x * blockDim.x + threadIdx.x;
  const int stride = gridDim.x * blockDim.x;
  for (; i < n4; i += stride) {
    float4 v = ((const float4*)src)[i];
    uint2 o;
    o.x = (u32)f2bf(v.x) | ((u32)f2bf(v.y) << 16);
    o.y = (u32)f2bf(v.z) | ((u32)f2bf(v.w) << 16);
    ((uint2*)dst)[i] = o;
  }
}

// ---------------------------------------------------------------------------
// Kernel 3: GEMM  C[m,n] = sum_k A[m,k] * W[n,k] + bias[n]   (torch Linear)
//   A: bf16 [M,K] row-major, W: bf16 [N,K] row-major (both K-contiguous).
//   MODE 0: store f32 [M,N]; MODE 1: store bf16 [M,N]; MODE 2: bf16 [N,M]^T.
//   Block: 8 waves, 128x128 tile; wave: 32x64 (2 M-tiles x 4 N-tiles).
//   K pipelined in 64-wide stages via async global->LDS, double buffered.
// ---------------------------------------------------------------------------
constexpr int GRS = 72;   // LDS row stride in u16 (64 data + 8 pad = 144 B)

template<int MODE>
__global__ __launch_bounds__(256) void gemm_bf16(const u16* __restrict__ A,
                                                 const u16* __restrict__ W,
                                                 const float* __restrict__ bias,
                                                 void* __restrict__ Cout,
                                                 int M, int N, int K) {
  __shared__ u16 smA[2][128 * GRS];   // 2 x 18 KB
  __shared__ u16 smB[2][128 * GRS];   // 2 x 18 KB

  const int t    = threadIdx.x;
  const int lane = t & 31;
  const int wid  = t >> 5;               // 0..7
  const int hl   = lane >> 4;            // half-wave 0/1
  const int ln   = lane & 15;
  const int wm   = wid >> 1;             // 0..3
  const int wn   = wid & 1;              // 0..1
  const int mblk = blockIdx.y * 128;
  const int nblk = blockIdx.x * 128;

  // per-thread stage-copy slots: 4 x 16B chunks covering 128 rows x 128 B
  int srow[4], scol[4];
  #pragma unroll
  for (int i = 0; i < 4; ++i) {
    const int j = t + i * 256;           // 0..1023
    srow[i] = j >> 3;
    scol[i] = (j & 7) * 8;               // u16 offset of 16B chunk
  }

  const v8f vzero = {0.f, 0.f, 0.f, 0.f, 0.f, 0.f, 0.f, 0.f};
  v8f acc[2][4];
  #pragma unroll
  for (int mt = 0; mt < 2; ++mt)
    #pragma unroll
    for (int nt = 0; nt < 4; ++nt) acc[mt][nt] = vzero;

  const int nstage = K / 64;

  auto issue_stage = [&](int ks, int buf) {
    const int kg = ks * 64;
    #pragma unroll
    for (int i = 0; i < 4; ++i)
      async_g2l_b128(A + (size_t)(mblk + srow[i]) * K + kg + scol[i],
                     &smA[buf][srow[i] * GRS + scol[i]]);
    #pragma unroll
    for (int i = 0; i < 4; ++i)
      async_g2l_b128(W + (size_t)(nblk + srow[i]) * K + kg + scol[i],
                     &smB[buf][srow[i] * GRS + scol[i]]);
  };

  issue_stage(0, 0);

  for (int ks = 0; ks < nstage; ++ks) {
    const int cur = ks & 1;
    if (ks + 1 < nstage) {
      issue_stage(ks + 1, cur ^ 1);
      __builtin_prefetch(A + (size_t)(mblk + srow[0]) * K + (ks + 2) * 64, 0, 1);
      __builtin_prefetch(W + (size_t)(nblk + srow[0]) * K + (ks + 2) * 64, 0, 1);
      wait_async<8>();                   // retire stage ks (in-order completion)
    } else {
      wait_async<0>();
    }
    __syncthreads();

    #pragma unroll
    for (int kk = 0; kk < 2; ++kk) {     // two 32-wide WMMA k-steps per stage
      Frag af[2], wf[4];
      #pragma unroll
      for (int mt = 0; mt < 2; ++mt) {
        // A layout: lane=M row; K chunks {hl*8 .. +7} and {hl*8+16 .. +7}
        const u16* p = &smA[cur][(wm * 32 + mt * 16 + ln) * GRS + kk * 32 + hl * 8];
        af[mt].q[0] = *(const uint4*)p;
        af[mt].q[1] = *(const uint4*)(p + 16);
      }
      #pragma unroll
      for (int nt = 0; nt < 4; ++nt) {
        // B layout: lane=N col; K chunks {hl*8 .. +7} and {hl*8+16 .. +7}
        const u16* p = &smB[cur][(wn * 64 + nt * 16 + ln) * GRS + kk * 32 + hl * 8];
        wf[nt].q[0] = *(const uint4*)p;
        wf[nt].q[1] = *(const uint4*)(p + 16);
      }
      #pragma unroll
      for (int mt = 0; mt < 2; ++mt)
        #pragma unroll
        for (int nt = 0; nt < 4; ++nt)
          acc[mt][nt] = wmma_bf16(af[mt].v, wf[nt].v, acc[mt][nt]);
    }
    __syncthreads();
  }

  // Epilogue. C layout: VGPR r -> row (hl*8 + r), col = ln.
  #pragma unroll
  for (int mt = 0; mt < 2; ++mt) {
    #pragma unroll
    for (int nt = 0; nt < 4; ++nt) {
      const int col  = nblk + wn * 64 + nt * 16 + ln;
      const int row0 = mblk + wm * 32 + mt * 16 + hl * 8;
      const float bs = bias[col];
      if (MODE == 0) {
        float* C = (float*)Cout;
        #pragma unroll
        for (int r = 0; r < 8; ++r)
          C[(size_t)(row0 + r) * N + col] = acc[mt][nt][r] + bs;
      } else if (MODE == 1) {
        u16* C = (u16*)Cout;
        #pragma unroll
        for (int r = 0; r < 8; ++r)
          C[(size_t)(row0 + r) * N + col] = f2bf(acc[mt][nt][r] + bs);
      } else {
        u16* C = (u16*)Cout;        // transposed: C[n, m], rows m contiguous
        Pack8 pk;
        #pragma unroll
        for (int r = 0; r < 8; ++r) pk.s[r] = f2bf(acc[mt][nt][r] + bs);
        *(uint4*)&C[(size_t)col * M + row0] = pk.q;
      }
    }
  }
}

// ---------------------------------------------------------------------------
// Kernel 4: masked attention for one (b, h, 32-query tile).
//   Qp: bf16 [B,S,D] (head-sliced via column offset), Kp: bf16 [B,P,D],
//   Vt: bf16 [D, B*P] (transposed so P·V B-operand is K(=key)-contiguous),
//   pidx: int [B,S], ctx out: bf16 [B,S,D].
//   Block: 64 threads (2 waves). LDS: 32 x 1032 f32 score rows (~129 KB).
// ---------------------------------------------------------------------------
constexpr int QT      = 32;     // query rows per block
constexpr int LDS_STR = 1032;   // f32 per score row (1024 + 8 pad)

__global__ __launch_bounds__(64) void attn_kernel(const u16* __restrict__ Qp,
                                                  const u16* __restrict__ Kp,
                                                  const u16* __restrict__ Vt,
                                                  const int* __restrict__ pidx,
                                                  u16* __restrict__ ctx) {
  __shared__ float sc[QT * LDS_STR];
  __shared__ int   spi[QT];

  const int nq = S_ / QT;                       // 128
  const int qt = blockIdx.x % nq;
  const int h  = (blockIdx.x / nq) % H_;
  const int b  = blockIdx.x / (nq * H_);
  const int q0 = qt * QT;

  const int lane = threadIdx.x & 31;
  const int w    = threadIdx.x >> 5;            // wave 0/1
  const int hl   = lane >> 4;
  const int ln   = lane & 15;
  const v8f vzero = {0.f, 0.f, 0.f, 0.f, 0.f, 0.f, 0.f, 0.f};

  if (threadIdx.x < QT)
    spi[threadIdx.x] = pidx[(size_t)b * S_ + q0 + threadIdx.x];
  __syncthreads();

  // ---------- phase 1: scores = (Q . K^T) * 1/sqrt(dh), masked ----------
  Frag qf[2][2];   // [M-tile][k-step] over 32 rows x dh=64
  #pragma unroll
  for (int mt = 0; mt < 2; ++mt)
    #pragma unroll
    for (int ks = 0; ks < 2; ++ks) {
      const u16* p = Qp + (size_t)(b * S_ + q0 + mt * 16 + ln) * D_
                        + h * DH_ + ks * 32 + hl * 8;
      qf[mt][ks].q[0] = *(const uint4*)p;
      qf[mt][ks].q[1] = *(const uint4*)(p + 16);
    }

  for (int kt = 0; kt < 32; ++kt) {             // each wave: 512 keys
    const int key0 = w * 512 + kt * 16;
    Frag kf[2];
    #pragma unroll
    for (int ks = 0; ks < 2; ++ks) {
      const u16* p = Kp + (size_t)(b * P_ + key0 + ln) * D_
                        + h * DH_ + ks * 32 + hl * 8;
      kf[ks].q[0] = *(const uint4*)p;
      kf[ks].q[1] = *(const uint4*)(p + 16);
    }
    #pragma unroll
    for (int mt = 0; mt < 2; ++mt) {
      v8f acc = vzero;
      acc = wmma_bf16(qf[mt][0].v, kf[0].v, acc);
      acc = wmma_bf16(qf[mt][1].v, kf[1].v, acc);
      const int colg = key0 + ln;               // global key index
      #pragma unroll
      for (int r = 0; r < 8; ++r) {
        const int row = mt * 16 + hl * 8 + r;   // local query row
        float v = acc[r] * 0.125f;              // 1/sqrt(64)
        if (colg > spi[row]) v = -1e30f;        // patch-boundary mask
        sc[row * LDS_STR + colg] = v;
      }
    }
  }
  __syncthreads();

  // ---------- phase 2: row softmax, in-place f32 -> bf16 probs ----------
  if (threadIdx.x < QT) {
    float* rp = &sc[threadIdx.x * LDS_STR];
    float m = -3.4e38f;
    for (int c = 0; c < P_; c += 4) {
      float4 v = *(const float4*)&rp[c];
      m = fmaxf(m, fmaxf(fmaxf(v.x, v.y), fmaxf(v.z, v.w)));
    }
    float s = 0.f;
    for (int c = 0; c < P_; c += 4) {
      float4 v = *(const float4*)&rp[c];
      s += __expf(v.x - m) + __expf(v.y - m) + __expf(v.z - m) + __expf(v.w - m);
    }
    const float inv = 1.0f / s;
    u16* bp = (u16*)rp;   // bf16 aliases front of row: write offset <= read
    for (int c = 0; c < P_; c += 4) {
      float4 v = *(const float4*)&rp[c];
      uint2 o;
      o.x = (u32)f2bf(__expf(v.x - m) * inv) | ((u32)f2bf(__expf(v.y - m) * inv) << 16);
      o.y = (u32)f2bf(__expf(v.z - m) * inv) | ((u32)f2bf(__expf(v.w - m) * inv) << 16);
      *(uint2*)&bp[c] = o;
    }
  }
  __syncthreads();

  // ---------- phase 3: ctx = probs . V (wave handles 16 rows x dh=64) -----
  const int rw = w * 16;
  v8f o[4];
  #pragma unroll
  for (int nt = 0; nt < 4; ++nt) o[nt] = vzero;

  const u16* sbase = (const u16*)sc;
  for (int k0 = 0; k0 < P_; k0 += 32) {
    Frag af;  // probs A-fragment from LDS (bf16 row stride = 2*LDS_STR)
    const u16* ap = sbase + (size_t)(rw + ln) * (2 * LDS_STR) + k0 + hl * 8;
    af.q[0] = *(const uint4*)ap;
    af.q[1] = *(const uint4*)(ap + 16);
    #pragma unroll
    for (int nt = 0; nt < 4; ++nt) {
      Frag vf;  // V^T B-fragment: key dim contiguous
      const u16* vp = Vt + (size_t)(h * DH_ + nt * 16 + ln) * (B_ * P_)
                         + b * P_ + k0 + hl * 8;
      vf.q[0] = *(const uint4*)vp;
      vf.q[1] = *(const uint4*)(vp + 16);
      o[nt] = wmma_bf16(af.v, vf.v, o[nt]);
    }
  }

  #pragma unroll
  for (int nt = 0; nt < 4; ++nt) {
    const int col  = h * DH_ + nt * 16 + ln;
    const int row0 = q0 + rw + hl * 8;
    #pragma unroll
    for (int r = 0; r < 8; ++r)
      ctx[(size_t)(b * S_ + row0 + r) * D_ + col] = f2bf(o[nt][r]);
  }
}

// ---------------------------------------------------------------------------
// Host-side launch
// ---------------------------------------------------------------------------
static constexpr size_t align_up(size_t x) { return (x + 255) & ~(size_t)255; }

extern "C" void kernel_launch(void* const* d_in, const int* in_sizes, int n_in,
                              void* d_out, int out_size, void* d_ws, size_t ws_size,
                              hipStream_t stream) {
  (void)in_sizes; (void)n_in; (void)out_size; (void)ws_size;

  const float* queries = (const float*)d_in[0];   // [B,S,D]
  const float* patches = (const float*)d_in[1];   // [B,P,D]
  const int*   bnd     = (const int*)  d_in[2];   // [B,S]
  const float* wq = (const float*)d_in[3];
  const float* wk = (const float*)d_in[4];
  const float* wv = (const float*)d_in[5];
  const float* wo = (const float*)d_in[6];
  const float* bq = (const float*)d_in[7];
  const float* bk = (const float*)d_in[8];
  const float* bv = (const float*)d_in[9];
  const float* bo = (const float*)d_in[10];

  // workspace layout
  char* ws = (char*)d_ws;
  size_t off = 0;
  auto take = [&](size_t bytes) { size_t o = off; off += align_up(bytes); return o; };
  const size_t o_pidx = take((size_t)B_ * S_ * 4);
  const size_t o_xq   = take((size_t)B_ * S_ * D_ * 2);
  const size_t o_xp   = take((size_t)B_ * P_ * D_ * 2);
  const size_t o_wq   = take((size_t)D_ * D_ * 2);
  const size_t o_wk   = take((size_t)D_ * D_ * 2);
  const size_t o_wv   = take((size_t)D_ * D_ * 2);
  const size_t o_wo   = take((size_t)D_ * D_ * 2);
  const size_t o_Qp   = take((size_t)B_ * S_ * D_ * 2);
  const size_t o_Kp   = take((size_t)B_ * P_ * D_ * 2);
  const size_t o_Vt   = take((size_t)D_ * B_ * P_ * 2);
  const size_t o_ctx  = take((size_t)B_ * S_ * D_ * 2);

  int* pidx = (int*)(ws + o_pidx);
  u16* xq   = (u16*)(ws + o_xq);
  u16* xp   = (u16*)(ws + o_xp);
  u16* wqb  = (u16*)(ws + o_wq);
  u16* wkb  = (u16*)(ws + o_wk);
  u16* wvb  = (u16*)(ws + o_wv);
  u16* wob  = (u16*)(ws + o_wo);
  u16* Qp   = (u16*)(ws + o_Qp);
  u16* Kp   = (u16*)(ws + o_Kp);
  u16* Vt   = (u16*)(ws + o_Vt);
  u16* ctx  = (u16*)(ws + o_ctx);

  // 1) patch-index cumsum
  cumsum_kernel<<<B_, 256, 0, stream>>>(bnd, pidx);

  // 2) f32 -> bf16 conversions
  cvt_bf16<<<1024, 256, 0, stream>>>(queries, xq, B_ * S_ * D_ / 4);
  cvt_bf16<<<1024, 256, 0, stream>>>(patches, xp, B_ * P_ * D_ / 4);
  cvt_bf16<<<512, 256, 0, stream>>>(wq, wqb, D_ * D_ / 4);
  cvt_bf16<<<512, 256, 0, stream>>>(wk, wkb, D_ * D_ / 4);
  cvt_bf16<<<512, 256, 0, stream>>>(wv, wvb, D_ * D_ / 4);
  cvt_bf16<<<512, 256, 0, stream>>>(wo, wob, D_ * D_ / 4);

  // 3) projections (Linear): Q [B*S,D], K [B*P,D], V^T [D, B*P]
  gemm_bf16<1><<<dim3(D_ / 128, (B_ * S_) / 128), 256, 0, stream>>>(
      xq, wqb, bq, Qp, B_ * S_, D_, D_);
  gemm_bf16<1><<<dim3(D_ / 128, (B_ * P_) / 128), 256, 0, stream>>>(
      xp, wkb, bk, Kp, B_ * P_, D_, D_);
  gemm_bf16<2><<<dim3(D_ / 128, (B_ * P_) / 128), 256, 0, stream>>>(
      xp, wvb, bv, Vt, B_ * P_, D_, D_);

  // 4) masked attention -> ctx bf16 [B,S,D]
  attn_kernel<<<B_ * H_ * (S_ / QT), 64, 0, stream>>>(Qp, Kp, Vt, pidx, ctx);

  // 5) output projection -> f32 d_out
  gemm_bf16<0><<<dim3(D_ / 128, (B_ * S_) / 128), 256, 0, stream>>>(
      ctx, wob, bo, (float*)d_out, B_ * S_, D_, D_);
}
